// SoftmaxStochasticPolicy_14534169329844
// MI455X (gfx1250) — compile-verified, matrix operand order checked by reading, not run on hardware
//
#include <hip/hip_runtime.h>

// ---------------------------------------------------------------------------
// logp[b] = dot(state[b], E[idx[b]]) - log( sum_n exp( dot(state[b], E[n]) ) )
// B=2048, D=64, N=100000, T=1.0
//
// V_WMMA_F32_16X16X32_F16 with split-f16 (hi+lo) emulation of f32:
//   s*e ~= s_hi*e_hi + s_hi*e_lo + s_lo*e_hi   (f32 accumulate inside WMMA)
// A (state) pre-scaled by log2(e) so v_exp_f32 (exp2) applies directly.
// Fast path: E pre-split to f16 hi/lo in WMMA-B-fragment layout in workspace,
// streamed to LDS with GLOBAL_LOAD_ASYNC_TO_LDS_B128 (double-buffered).
// ---------------------------------------------------------------------------

typedef __attribute__((ext_vector_type(16))) _Float16 v16h;
typedef __attribute__((ext_vector_type(8)))  float    v8f;
typedef int v4i __attribute__((vector_size(16)));   // matches builtin proto

union AFrag { v16h v; _Float16 e[16]; };
union CFrag { v8f  v; float     e[8];  };

__device__ __forceinline__ v8f wmma_f16(v16h a, v16h b, v8f c) {
  return __builtin_amdgcn_wmma_f32_16x16x32_f16(false, a, false, b,
                                                (short)0, c, false, false);
}

// 16-bit fragment mapping (ISA 7.12.2):
//   lane = idx%16 + 16*((kk%16)>=8); vgpr = 4*(kk/16)+(kk%8)/2; half = kk%2

#if defined(__gfx1250__) && __has_builtin(__builtin_amdgcn_global_load_async_to_lds_b128)
#define USE_ASYNC 1
#else
#define USE_ASYNC 0
#endif

#if USE_ASYNC
typedef __attribute__((address_space(1))) v4i glob_v4i;
typedef __attribute__((address_space(3))) v4i lds_v4i;

template <int OFF>
__device__ __forceinline__ void async_cp16(void* lds, const void* g) {
  __builtin_amdgcn_global_load_async_to_lds_b128(
      (glob_v4i*)(__UINTPTR_TYPE__)(g),
      (lds_v4i*)(__UINTPTR_TYPE__)(lds),
      OFF, 0);
}
#endif

// ---------------------------------------------------------------------------
// Pre-pass: split E into f16 hi/lo, stored per 16-col tile in B-fragment
// order. Tile t occupies 4KB: hi halfs [kchunk][lane][16] then lo halfs.
// ---------------------------------------------------------------------------
__global__ __launch_bounds__(256) void split_prepass_kernel(
    const float* __restrict__ emb, _Float16* __restrict__ wsE)
{
  const int t   = blockIdx.x;
  const int tid = threadIdx.x;
  const int sn  = tid >> 4;
  const int sk4 = (tid & 15) * 4;

  const float4 q = *(const float4*)(emb + (size_t)(t * 16 + sn) * 64 + sk4);
  const float xv[4] = {q.x, q.y, q.z, q.w};

  _Float16* wt = wsE + (size_t)t * 2048;   // 2048 halfs per tile
#pragma unroll
  for (int j = 0; j < 4; ++j) {
    const int k  = sk4 + j;
    const int c  = k >> 5;
    const int kk = k & 31;
    const int hs = (kk & 15) >> 3;
    const int v  = 4 * (kk >> 4) + ((kk & 7) >> 1);
    const int h  = kk & 1;
    const int lane = sn + 16 * hs;
    const int idx  = (c * 32 + lane) * 16 + 2 * v + h;
    const float x = xv[j];
    const _Float16 fh = (_Float16)x;
    wt[idx]        = fh;                        // hi region (first 1024 halfs)
    wt[idx + 1024] = (_Float16)(x - (float)fh); // lo region
  }
}

// ---------------------------------------------------------------------------
// Fast main kernel: 256 rows/block (8 waves x 2 row-blocks x 16 rows),
// async double-buffered LDS staging, no masks (exact tiling required).
// ---------------------------------------------------------------------------
__global__ __launch_bounds__(256) void fast_sumexp_kernel(
    const float* __restrict__ state,   // [B][64]
    const char*  __restrict__ wsE,     // split E, 4KB per tile
    float* __restrict__ partial,       // [B][nch]
    int tilesPerChunk)
{
  __shared__ __align__(32) char sB[2][8192];   // double-buffered tile pair

  const int tid   = threadIdx.x;
  const int lane  = tid & 31;
  const int wave  = tid >> 5;
  const int chunk = blockIdx.x;
  const int nch   = gridDim.x;
  const int t0    = chunk * tilesPerChunk;
  const int t1    = t0 + tilesPerChunk;        // even count, all tiles full

  const float LOG2E = 1.44269504088896341f;

  // A fragments for 2 row-blocks (hi/lo split, pre-scaled by log2e).
  AFrag aHi[2][2], aLo[2][2];                  // [rowblock][kchunk]
#pragma unroll
  for (int rb = 0; rb < 2; ++rb) {
    const int m  = lane & 15;
    const int hs = lane >> 4;
    const int row = blockIdx.y * 256 + (wave + rb * 8) * 16 + m;
    const float* srow = state + (size_t)row * 64;
#pragma unroll
    for (int c = 0; c < 2; ++c)
#pragma unroll
      for (int v = 0; v < 8; ++v)
#pragma unroll
        for (int h = 0; h < 2; ++h) {
          const int kk = 16 * (v >> 2) + 8 * hs + 2 * (v & 3) + h;
          const float x = srow[c * 32 + kk] * LOG2E;
          const _Float16 fh = (_Float16)x;
          aHi[rb][c].e[2 * v + h] = fh;
          aLo[rb][c].e[2 * v + h] = (_Float16)(x - (float)fh);
        }
  }

  const char* srcBase = wsE + (size_t)t0 * 4096 + tid * 32;  // this thread's 32B
  float sums[2][8];
#pragma unroll
  for (int rb = 0; rb < 2; ++rb)
#pragma unroll
    for (int i = 0; i < 8; ++i) sums[rb][i] = 0.f;

  // Prologue: stage first pair into buffer 0.
  {
    char* d = &sB[0][0] + tid * 32;
#if USE_ASYNC
    async_cp16<0>(d, srcBase);
    async_cp16<16>(d, srcBase);
    asm volatile("s_wait_asynccnt 0x0" ::: "memory");
#else
    ((int4*)d)[0] = ((const int4*)srcBase)[0];
    ((int4*)d)[1] = ((const int4*)srcBase)[1];
#endif
  }
  __syncthreads();

  int pb = 0;
  const char* srcN = srcBase + 8192;
#pragma unroll 1
  for (int t = t0; t < t1; t += 2) {
    // Prefetch next pair into the other buffer (clamped, uniform select).
    const char* s = (t + 2 < t1) ? srcN : srcBase;
    char* d = &sB[pb ^ 1][0] + tid * 32;
#if USE_ASYNC
    async_cp16<0>(d, s);
    async_cp16<16>(d, s);
#else
    const int4 r0 = ((const int4*)s)[0];
    const int4 r1 = ((const int4*)s)[1];
#endif
    srcN += 8192;

    const char* base = &sB[pb][0] + (size_t)lane * 32;

    CFrag a00, a01, a10, a11;   // [rowblock][tile]

    // ---- tile 0 ----
    {
      const v16h bh0 = *(const v16h*)(base + 0);
      const v16h bh1 = *(const v16h*)(base + 1024);
      const v16h bl0 = *(const v16h*)(base + 2048);
      const v16h bl1 = *(const v16h*)(base + 3072);
#pragma unroll
      for (int i = 0; i < 8; ++i) { a00.e[i] = 0.f; a10.e[i] = 0.f; }
      a00.v = wmma_f16(aHi[0][0].v, bh0, a00.v);
      a10.v = wmma_f16(aHi[1][0].v, bh0, a10.v);
      a00.v = wmma_f16(aHi[0][0].v, bl0, a00.v);
      a10.v = wmma_f16(aHi[1][0].v, bl0, a10.v);
      a00.v = wmma_f16(aLo[0][0].v, bh0, a00.v);
      a10.v = wmma_f16(aLo[1][0].v, bh0, a10.v);
      a00.v = wmma_f16(aHi[0][1].v, bh1, a00.v);
      a10.v = wmma_f16(aHi[1][1].v, bh1, a10.v);
      a00.v = wmma_f16(aHi[0][1].v, bl1, a00.v);
      a10.v = wmma_f16(aHi[1][1].v, bl1, a10.v);
      a00.v = wmma_f16(aLo[0][1].v, bh1, a00.v);
      a10.v = wmma_f16(aLo[1][1].v, bh1, a10.v);
    }
    // ---- tile 1 ----
    {
      const v16h bh0 = *(const v16h*)(base + 4096);
      const v16h bh1 = *(const v16h*)(base + 5120);
      const v16h bl0 = *(const v16h*)(base + 6144);
      const v16h bl1 = *(const v16h*)(base + 7168);
#pragma unroll
      for (int i = 0; i < 8; ++i) { a01.e[i] = 0.f; a11.e[i] = 0.f; }
      a01.v = wmma_f16(aHi[0][0].v, bh0, a01.v);
      a11.v = wmma_f16(aHi[1][0].v, bh0, a11.v);
      a01.v = wmma_f16(aHi[0][0].v, bl0, a01.v);
      a11.v = wmma_f16(aHi[1][0].v, bl0, a11.v);
      a01.v = wmma_f16(aLo[0][0].v, bh0, a01.v);
      a11.v = wmma_f16(aLo[1][0].v, bh0, a11.v);
      a01.v = wmma_f16(aHi[0][1].v, bh1, a01.v);
      a11.v = wmma_f16(aHi[1][1].v, bh1, a11.v);
      a01.v = wmma_f16(aHi[0][1].v, bl1, a01.v);
      a11.v = wmma_f16(aHi[1][1].v, bl1, a11.v);
      a01.v = wmma_f16(aLo[0][1].v, bh1, a01.v);
      a11.v = wmma_f16(aLo[1][1].v, bh1, a11.v);
    }

    // exp2 accumulation (A pre-scaled by log2e).
#pragma unroll
    for (int i = 0; i < 8; ++i) {
      sums[0][i] += __builtin_amdgcn_exp2f(a00.e[i]) +
                    __builtin_amdgcn_exp2f(a01.e[i]);
      sums[1][i] += __builtin_amdgcn_exp2f(a10.e[i]) +
                    __builtin_amdgcn_exp2f(a11.e[i]);
    }

#if USE_ASYNC
    asm volatile("s_wait_asynccnt 0x0" ::: "memory");
#else
    ((int4*)d)[0] = r0;
    ((int4*)d)[1] = r1;
#endif
    __syncthreads();
    pb ^= 1;
  }

  // Reduce across the 16 lanes sharing each row; write partials.
#pragma unroll
  for (int rb = 0; rb < 2; ++rb) {
#pragma unroll
    for (int off = 8; off >= 1; off >>= 1)
#pragma unroll
      for (int i = 0; i < 8; ++i)
        sums[rb][i] += __shfl_xor(sums[rb][i], off, 32);
    if ((lane & 15) == 0) {
      const int r0 = blockIdx.y * 256 + (wave + rb * 8) * 16 + (lane >> 4) * 8;
#pragma unroll
      for (int i = 0; i < 8; ++i)
        partial[(size_t)(r0 + i) * nch + chunk] = sums[rb][i];
    }
  }
}

// ---------------------------------------------------------------------------
// General fallback kernel (any N; B%128==0): in-loop conversion, masked tail.
// ---------------------------------------------------------------------------
typedef _Float16 BStage[2][2][32][16];
struct Stg { float v[2][4]; };

__device__ __forceinline__ Stg stage_load(const float* __restrict__ emb,
                                          int t, int t1, int N, int sn, int sk4) {
  Stg s;
#pragma unroll
  for (int tt = 0; tt < 2; ++tt) {
    const int tg  = t + tt;
    const int col = tg * 16 + sn;
    int colc = col; if (colc > N - 1) colc = N - 1;
    const float4 q = *(const float4*)(emb + (size_t)colc * 64 + sk4);
    const float m = ((tg < t1) && (col < N)) ? 1.f : 0.f;
    s.v[tt][0] = q.x * m; s.v[tt][1] = q.y * m;
    s.v[tt][2] = q.z * m; s.v[tt][3] = q.w * m;
  }
  return s;
}

__device__ __forceinline__ void stage_write(BStage& hi, BStage& lo,
                                            const Stg& s, int sn, int sk4) {
#pragma unroll
  for (int tt = 0; tt < 2; ++tt)
#pragma unroll
    for (int j = 0; j < 4; ++j) {
      const int k  = sk4 + j;
      const int c  = k >> 5;
      const int kk = k & 31;
      const int hs = (kk & 15) >> 3;
      const int v  = 4 * (kk >> 4) + ((kk & 7) >> 1);
      const int h  = kk & 1;
      const float x = s.v[tt][j];
      const _Float16 fh = (_Float16)x;
      hi[tt][c][sn + 16 * hs][2 * v + h] = fh;
      lo[tt][c][sn + 16 * hs][2 * v + h] = (_Float16)(x - (float)fh);
    }
}

__global__ __launch_bounds__(256) void general_sumexp_kernel(
    const float* __restrict__ state, const float* __restrict__ emb,
    float* __restrict__ partial, int N, int nTiles, int tilesPerChunk)
{
  __shared__ __align__(32) BStage sBhi[2];
  __shared__ __align__(32) BStage sBlo[2];

  const int tid = threadIdx.x, lane = tid & 31, wave = tid >> 5;
  const int rowBase = blockIdx.y * 128 + wave * 16;
  const int chunk = blockIdx.x, nch = gridDim.x;
  const int t0 = chunk * tilesPerChunk;
  int t1 = t0 + tilesPerChunk; if (t1 > nTiles) t1 = nTiles;

  const float LOG2E = 1.44269504088896341f;
  AFrag aHi[2], aLo[2];
  {
    const int m = lane & 15, hs = lane >> 4;
    const float* srow = state + (size_t)(rowBase + m) * 64;
#pragma unroll
    for (int c = 0; c < 2; ++c)
#pragma unroll
      for (int v = 0; v < 8; ++v)
#pragma unroll
        for (int h = 0; h < 2; ++h) {
          const int kk = 16 * (v >> 2) + 8 * hs + 2 * (v & 3) + h;
          const float x = srow[c * 32 + kk] * LOG2E;
          const _Float16 fh = (_Float16)x;
          aHi[c].e[2 * v + h] = fh;
          aLo[c].e[2 * v + h] = (_Float16)(x - (float)fh);
        }
  }

  const int sn = tid >> 4, sk4 = (tid & 15) * 4;
  float sums[8] = {0.f,0.f,0.f,0.f,0.f,0.f,0.f,0.f};

  if (t0 < t1) {
    Stg cur = stage_load(emb, t0, t1, N, sn, sk4);
    stage_write(sBhi[0], sBlo[0], cur, sn, sk4);
    __syncthreads();
    int pb = 0;
#pragma unroll 1
    for (int t = t0; t < t1; t += 2) {
      Stg nxt = stage_load(emb, t + 2, t1, N, sn, sk4);
      const v16h bh00 = *(const v16h*)(&sBhi[pb][0][0][lane][0]);
      const v16h bh01 = *(const v16h*)(&sBhi[pb][0][1][lane][0]);
      const v16h bl00 = *(const v16h*)(&sBlo[pb][0][0][lane][0]);
      const v16h bl01 = *(const v16h*)(&sBlo[pb][0][1][lane][0]);
      const v16h bh10 = *(const v16h*)(&sBhi[pb][1][0][lane][0]);
      const v16h bh11 = *(const v16h*)(&sBhi[pb][1][1][lane][0]);
      const v16h bl10 = *(const v16h*)(&sBlo[pb][1][0][lane][0]);
      const v16h bl11 = *(const v16h*)(&sBlo[pb][1][1][lane][0]);
      CFrag a0, a1;
#pragma unroll
      for (int i = 0; i < 8; ++i) { a0.e[i] = 0.f; a1.e[i] = 0.f; }
      a0.v = wmma_f16(aHi[0].v, bh00, a0.v);
      a1.v = wmma_f16(aHi[0].v, bh10, a1.v);
      a0.v = wmma_f16(aHi[0].v, bl00, a0.v);
      a1.v = wmma_f16(aHi[0].v, bl10, a1.v);
      a0.v = wmma_f16(aLo[0].v, bh00, a0.v);
      a1.v = wmma_f16(aLo[0].v, bh10, a1.v);
      a0.v = wmma_f16(aHi[1].v, bh01, a0.v);
      a1.v = wmma_f16(aHi[1].v, bh11, a1.v);
      a0.v = wmma_f16(aHi[1].v, bl01, a0.v);
      a1.v = wmma_f16(aHi[1].v, bl11, a1.v);
      a0.v = wmma_f16(aLo[1].v, bh01, a0.v);
      a1.v = wmma_f16(aLo[1].v, bh11, a1.v);
#pragma unroll
      for (int i = 0; i < 8; ++i)
        sums[i] += __builtin_amdgcn_exp2f(a0.e[i]) +
                   __builtin_amdgcn_exp2f(a1.e[i]);
      stage_write(sBhi[pb ^ 1], sBlo[pb ^ 1], nxt, sn, sk4);
      __syncthreads();
      pb ^= 1;
    }
  }
  {
    float pads = 0.f;
    if (t0 < t1) {
      const int nproc = (((t1 - t0) + 1) >> 1) * 2;
      pads = (float)((t0 + nproc) - t1);
      const int rem = N & 15;
      if (rem && (nTiles - 1) >= t0 && (nTiles - 1) < t1 && (lane & 15) >= rem)
        pads += 1.f;
    }
#pragma unroll
    for (int i = 0; i < 8; ++i) sums[i] -= pads;
  }
#pragma unroll
  for (int off = 8; off >= 1; off >>= 1)
#pragma unroll
    for (int i = 0; i < 8; ++i)
      sums[i] += __shfl_xor(sums[i], off, 32);
  if ((lane & 15) == 0) {
    const int r0 = rowBase + (lane >> 4) * 8;
#pragma unroll
    for (int i = 0; i < 8; ++i)
      partial[(size_t)(r0 + i) * nch + chunk] = sums[i];
  }
}

// One wave per row: reduce chunk partials, exact-f32 selected dot, finalize.
__global__ __launch_bounds__(256) void finalize_kernel(
    const float* __restrict__ state,
    const int*   __restrict__ item_index,
    const float* __restrict__ emb,
    const float* __restrict__ partial,
    float* __restrict__ out, int B, int nch)
{
  const int lane = threadIdx.x & 31;
  const int wave = threadIdx.x >> 5;
  const int row  = blockIdx.x * 8 + wave;
  if (row >= B) return;

  float s = 0.f;
  for (int c = lane; c < nch; c += 32)
    s += partial[(size_t)row * nch + c];

  const int it = item_index[row];
  const int d  = lane * 2;
  float dot = state[(size_t)row * 64 + d]     * emb[(size_t)it * 64 + d] +
              state[(size_t)row * 64 + d + 1] * emb[(size_t)it * 64 + d + 1];

#pragma unroll
  for (int off = 16; off >= 1; off >>= 1) {
    s   += __shfl_xor(s, off, 32);
    dot += __shfl_xor(dot, off, 32);
  }
  if (lane == 0)
    out[row] = dot - __builtin_amdgcn_logf(s) * 0.693147180559945f;
}

extern "C" void kernel_launch(void* const* d_in, const int* in_sizes, int n_in,
                              void* d_out, int out_size, void* d_ws, size_t ws_size,
                              hipStream_t stream) {
  const float* state = (const float*)d_in[0];
  const int*   index = (const int*)d_in[1];
  const float* emb   = (const float*)d_in[2];
  float*       out   = (float*)d_out;

  const int B = in_sizes[1];           // 2048
  const int N = in_sizes[2] / 64;      // 100000
  const int nTiles = (N + 15) / 16;    // 6250

  const int NCHF = 125;                // fast path: 125 chunks x 50 tiles
  const size_t SE = (size_t)nTiles * 4096;
  const bool fast = (N % 16 == 0) && (B % 256 == 0) &&
                    (nTiles % (2 * NCHF) == 0) &&
                    (ws_size >= SE + (size_t)B * NCHF * sizeof(float));

  if (fast) {
    float* partial = (float*)((char*)d_ws + SE);
    split_prepass_kernel<<<nTiles, 256, 0, stream>>>(emb, (_Float16*)d_ws);
    dim3 grid(NCHF, B / 256);
    fast_sumexp_kernel<<<grid, 256, 0, stream>>>(state, (const char*)d_ws,
                                                 partial, nTiles / NCHF);
    finalize_kernel<<<(B + 7) / 8, 256, 0, stream>>>(state, index, emb,
                                                     partial, out, B, NCHF);
  } else {
    const int NCH = 50;
    const int tpc = (nTiles + NCH - 1) / NCH;
    float* partial = (float*)d_ws;
    dim3 grid(NCH, B / 128);
    general_sumexp_kernel<<<grid, 256, 0, stream>>>(state, emb, partial,
                                                    N, nTiles, tpc);
    finalize_kernel<<<(B + 7) / 8, 256, 0, stream>>>(state, index, emb,
                                                     partial, out, B, NCH);
  }
}
